// VanillaRNN_77180562309326
// MI455X (gfx1250) — compile-verified, matrix operand order checked by reading, not run on hardware
//
#include <hip/hip_runtime.h>
#include <stdint.h>

// ---- problem dims (from reference) ----
#define SEQ_LEN 128
#define IN_DIM  512
#define HID     2048
#define OUT_DIM 10
#define BATCH   256

// ---- tiling ----
#define KC      32              // K elements per chunk (one WMMA K-step)
#define NCX     (IN_DIM / KC)   // 16 chunks for x projection
#define NCH     (HID / KC)      // 64 chunks for h recurrence
#define WG_M    32              // batch rows per workgroup
#define WG_N    128             // hidden cols per workgroup
#define ROWB    80              // padded bytes per 32-elem bf16 row (64B data + 16B pad)
#define A_BYTES (WG_M * ROWB)           // 2560
#define B_BYTES (WG_N * ROWB)           // 10240
#define BUF_BYTES (A_BYTES + B_BYTES)   // 12800 per buffer, x2 double buffered

typedef __attribute__((ext_vector_type(16))) __bf16 v16bf;
typedef __attribute__((ext_vector_type(8)))  float  v8f;

union FragBF {
    v16bf v;
    uint4 q[2];
};

// ---- CDNA5 async global->LDS copy (ASYNCcnt path), with safe fallback ----
#if __has_builtin(__builtin_amdgcn_global_load_async_to_lds_b128)
#define HAVE_ASYNC 1
typedef int vec4i __attribute__((vector_size(16)));
typedef __attribute__((address_space(1))) vec4i* gv4i_ptr;  // global (AS1)
typedef __attribute__((address_space(3))) vec4i* lv4i_ptr;  // LDS (AS3)
__device__ __forceinline__ void copy16(const void* g, void* l) {
    __builtin_amdgcn_global_load_async_to_lds_b128((gv4i_ptr)g, (lv4i_ptr)l, 0, 0);
}
#if __has_builtin(__builtin_amdgcn_s_wait_asynccnt)
#define WAIT_ASYNC(n) __builtin_amdgcn_s_wait_asynccnt(n)
#else
#define WAIT_ASYNC(n) asm volatile("s_wait_asynccnt %0" ::"i"(n) : "memory")
#endif
#else
#define HAVE_ASYNC 0
__device__ __forceinline__ void copy16(const void* g, void* l) {
    *(uint4*)l = *(const uint4*)g;   // sync fallback: global_load_b128 + ds_store_b128
}
#define WAIT_ASYNC(n)
#endif

// scheduling hints: group DS-reads ahead of the WMMA chain
#if __has_builtin(__builtin_amdgcn_sched_group_barrier)
#define SCHED_GROUP(mask, size, id) __builtin_amdgcn_sched_group_barrier(mask, size, id)
#else
#define SCHED_GROUP(mask, size, id)
#endif

__device__ __forceinline__ unsigned short f32_to_bf16(float f) {
    unsigned u = __builtin_bit_cast(unsigned, f);
    u += 0x7FFFu + ((u >> 16) & 1u);   // round-to-nearest-even
    return (unsigned short)(u >> 16);
}

__device__ __forceinline__ float bf16_to_f32(unsigned short h) {
    unsigned u = ((unsigned)h) << 16;
    return __builtin_bit_cast(float, u);
}

// ---------------- fp32 -> bf16 conversion (one-time prep) ----------------
__global__ void cvt_f32_bf16_kernel(const float* __restrict__ in,
                                    unsigned short* __restrict__ out, int n) {
    int i = blockIdx.x * blockDim.x + threadIdx.x;
    if (i < n) out[i] = f32_to_bf16(in[i]);
}

// ---------------- one recurrence step ----------------
// h_out[m,n] = tanh( x_t[m,:]·Whx[n,:] + bhx[n] (+ hprev[m,:]·Whh[n,:] + bhh[n]) )
// WG: 128 threads = 4 waves (2M x 2N); wave tile 16(M) x 64(N) = 4 wmma / K-step.
// A+B tiles staged in LDS, double buffered via async copies.
__global__ void __launch_bounds__(128, 1)
rnn_step_kernel(const unsigned short* __restrict__ xt, int x_stride,
                const unsigned short* __restrict__ Whx,
                const float* __restrict__ bhx,
                const unsigned short* __restrict__ hprev,
                const unsigned short* __restrict__ Whh,
                const float* __restrict__ bhh,
                unsigned short* __restrict__ hout,
                int use_hh)
{
    __shared__ __attribute__((aligned(16))) unsigned char smem[2][BUF_BYTES];

    const int tid  = threadIdx.x;
    const int lane = tid & 31;
    const int wave = tid >> 5;          // 0..3
    const int mw   = wave >> 1;         // 0..1
    const int nw   = wave & 1;          // 0..1
    const int m_blk = blockIdx.x * WG_M;
    const int n_blk = blockIdx.y * WG_N;
    const int r15 = lane & 15;
    const int hi  = lane >> 4;          // lane half (K sub-chunk select, ISA layout)
    const int c0  = hi * 8;

    const int NC = NCX + (use_hh ? NCH : 0);

    // stage one 32-wide K chunk of A (32 rows) and B (128 rows) into buf
    auto issue = [&](int c, unsigned char* buf) {
        const unsigned short* asrc; int astr;
        const unsigned short* bsrc; int bstr; int k;
        if (c < NCX) { asrc = xt;    astr = x_stride; bsrc = Whx; bstr = IN_DIM; k = c * KC; }
        else         { asrc = hprev; astr = HID;      bsrc = Whh; bstr = HID;    k = (c - NCX) * KC; }
        {   // A: 32 rows x 4 x 16B chunks = 128 -> one per thread
            const int row = tid >> 2, ch = tid & 3;
            copy16(asrc + (size_t)(m_blk + row) * astr + k + ch * 8,
                   buf + row * ROWB + ch * 16);
        }
#pragma unroll
        for (int j = 0; j < 4; ++j) {   // B: 128 rows x 4 chunks = 512 -> four per thread
            const int idx = tid + j * 128;
            const int row = idx >> 2, ch = idx & 3;
            copy16(bsrc + (size_t)(n_blk + row) * bstr + k + ch * 8,
                   buf + A_BYTES + row * ROWB + ch * 16);
        }
    };

    v8f acc[4] = {};

    issue(0, smem[0]);
    for (int c = 0; c < NC; ++c) {
        const int p = c & 1;
        if (c + 1 < NC) {
            issue(c + 1, smem[p ^ 1]);   // prefetch next chunk into other buffer
            WAIT_ASYNC(5);               // 5 outstanding allowed = next chunk's copies
        } else {
            WAIT_ASYNC(0);
        }
        __syncthreads();                 // all waves' chunk-c copies visible

        // issue ALL fragment ds_loads first, then the 4-wmma chain.
        const unsigned char* bufp = smem[p];
        FragBF a;
        const unsigned char* arow = bufp + (mw * 16 + r15) * ROWB;
        a.q[0] = *(const uint4*)(arow + c0 * 2);
        a.q[1] = *(const uint4*)(arow + 32 + c0 * 2);
        FragBF b[4];
#pragma unroll
        for (int nt = 0; nt < 4; ++nt) {
            const unsigned char* brow = bufp + A_BYTES + (nw * 64 + nt * 16 + r15) * ROWB;
            b[nt].q[0] = *(const uint4*)(brow + c0 * 2);
            b[nt].q[1] = *(const uint4*)(brow + 32 + c0 * 2);
        }
#pragma unroll
        for (int nt = 0; nt < 4; ++nt) {
            acc[nt] = __builtin_amdgcn_wmma_f32_16x16x32_bf16(
                false, a.v, false, b[nt].v, (short)0, acc[nt], false, false);
        }
        // pin the schedule: 10 DS-reads (mask 0x200) first, then 4 WMMA (0x8)
        SCHED_GROUP(0x200, 10, 0);
        SCHED_GROUP(0x008, 4, 0);
        __syncthreads();                 // done reading buf p before it is refilled
    }

    // ---- epilogue: bias + tanh, store bf16 ----
    // D layout: VGPR r, lanes 0-15 -> M=r, N=lane; lanes 16-31 -> M=8+r, N=lane-16.
    const int m_base = m_blk + mw * 16;
    const int n_base = n_blk + nw * 64;
#pragma unroll
    for (int nt = 0; nt < 4; ++nt) {
        const int n = n_base + nt * 16 + r15;
        const float bias = bhx[n] + (use_hh ? bhh[n] : 0.0f);
#pragma unroll
        for (int r = 0; r < 8; ++r) {
            const int m = m_base + hi * 8 + r;
            float v = acc[nt][r] + bias;
            v = tanhf(v);
            hout[(size_t)m * HID + n] = f32_to_bf16(v);
        }
    }
}

// ---------------- output projection: [256,2048] x [2048,10]^T ----------------
__global__ void out_proj_kernel(const unsigned short* __restrict__ h,
                                const float* __restrict__ Wout,
                                const float* __restrict__ bout,
                                float* __restrict__ out)
{
    const int b = threadIdx.x;   // 0..255
    const int o = blockIdx.x;    // 0..9
    const unsigned short* hrow = h + (size_t)b * HID;
    const float* wrow = Wout + (size_t)o * HID;
    float acc = 0.0f;
    for (int k = 0; k < HID; ++k)
        acc += bf16_to_f32(hrow[k]) * wrow[k];
    out[b * OUT_DIM + o] = acc + bout[o];
}

extern "C" void kernel_launch(void* const* d_in, const int* in_sizes, int n_in,
                              void* d_out, int out_size, void* d_ws, size_t ws_size,
                              hipStream_t stream)
{
    const float* x     = (const float*)d_in[0];
    const float* W_hx  = (const float*)d_in[1];
    const float* b_hx  = (const float*)d_in[2];
    const float* W_hh  = (const float*)d_in[3];
    const float* b_hh  = (const float*)d_in[4];
    const float* W_out = (const float*)d_in[5];
    const float* b_out = (const float*)d_in[6];
    float* out = (float*)d_out;

    // workspace layout (bf16, ~46 MB total)
    unsigned short* Whh_bf = (unsigned short*)d_ws;                  // 2048*2048
    unsigned short* Whx_bf = Whh_bf + (size_t)HID * HID;             // 2048*512
    unsigned short* hbuf0  = Whx_bf + (size_t)HID * IN_DIM;          // 256*2048
    unsigned short* hbuf1  = hbuf0  + (size_t)BATCH * HID;           // 256*2048
    unsigned short* x_bf   = hbuf1  + (size_t)BATCH * HID;           // 256*128*512

    // one-time fp32 -> bf16 conversion of weights and inputs
    {
        int n = HID * HID;
        cvt_f32_bf16_kernel<<<(n + 255) / 256, 256, 0, stream>>>(W_hh, Whh_bf, n);
        n = HID * IN_DIM;
        cvt_f32_bf16_kernel<<<(n + 255) / 256, 256, 0, stream>>>(W_hx, Whx_bf, n);
        n = BATCH * SEQ_LEN * IN_DIM;
        cvt_f32_bf16_kernel<<<(n + 255) / 256, 256, 0, stream>>>(x, x_bf, n);
    }

    dim3 grid(BATCH / WG_M, HID / WG_N);   // (8, 16) = 128 WGs

    // t = 0: no hh branch (matches reference h0 = tanh(xw[:,0]))
    rnn_step_kernel<<<grid, 128, 0, stream>>>(
        x_bf /* t=0 */, SEQ_LEN * IN_DIM, Whx_bf, b_hx,
        nullptr, nullptr, nullptr, hbuf0, 0);

    unsigned short* bufs[2] = { hbuf0, hbuf1 };
    for (int t = 1; t <= SEQ_LEN - 2; ++t) {
        unsigned short* hin = bufs[(t - 1) & 1];
        unsigned short* hou = bufs[t & 1];
        rnn_step_kernel<<<grid, 128, 0, stream>>>(
            x_bf + (size_t)t * IN_DIM, SEQ_LEN * IN_DIM, Whx_bf, b_hx,
            hin, Whh_bf, b_hh, hou, 1);
    }

    unsigned short* hlast = bufs[(SEQ_LEN - 2) & 1];  // t=126 -> hbuf0
    out_proj_kernel<<<OUT_DIM, BATCH, 0, stream>>>(hlast, W_out, b_out, out);
}
